// TextureDrivenWatermark_67327907332295
// MI455X (gfx1250) — matrix operand order repaired and checked by reading, not compiled, not run on hardware
//
#include <hip/hip_runtime.h>
#include <hip/hip_bf16.h>
#include <math.h>

// ---------------------------------------------------------------------------
// TextureDrivenWatermark — CDNA5 (gfx1250) implementation.
// DCT/IDCT pair collapsed to rank-1 update; 7x7x4 conv done with
// v_wmma_f32_16x16x32_bf16 (implicit GEMM, M=16 pixels, K=224 padded).
// Conv patch staged into LDS via the Tensor Data Mover (tensor_load_to_lds)
// when the builtin is available; cooperative-load fallback otherwise.
// ---------------------------------------------------------------------------

typedef __attribute__((ext_vector_type(16))) __bf16 v16bf;
typedef __attribute__((ext_vector_type(8)))  float  v8f;

#if defined(__has_builtin)
#if __has_builtin(__builtin_amdgcn_tensor_load_to_lds) && \
    __has_builtin(__builtin_amdgcn_s_wait_tensorcnt)
#define HAVE_TDM 1
#endif
#endif

#ifdef HAVE_TDM
typedef __attribute__((ext_vector_type(4))) unsigned int u32x4;
typedef __attribute__((ext_vector_type(8))) int          i32x8;
typedef __attribute__((ext_vector_type(4))) int          i32x4;
#endif

#define BATCH 16
#define HH    512
#define WW    512
#define HWPX  (HH * WW)
#define WP    518          // 512 + 2*3 zero pad for 'SAME' 7x7 conv
#define WP2   (WP * WP)
#define ALPHA_C 0.05f

// Row 4 and row 3 of the orthonormal 8-point DCT-II matrix.
__device__ __constant__ float c_D4[8] = {
    0.35355339f, -0.35355339f, -0.35355339f,  0.35355339f,
    0.35355339f, -0.35355339f, -0.35355339f,  0.35355339f };
__device__ __constant__ float c_D3[8] = {
    0.41573481f, -0.09754516f, -0.49039264f, -0.27778512f,
    0.27778512f,  0.49039264f,  0.09754516f, -0.41573481f };

__device__ __forceinline__ int iclamp(int v, int lo, int hi) {
    return v < lo ? lo : (v > hi ? hi : v);
}
__device__ __forceinline__ float sigmoidf(float x) {
    return 1.0f / (1.0f + __expf(-x));
}

// ---------------------------------------------------------------------------
// K1: gray, per-image YUV sums, bf16 RGB into padded conv-input buffer.
// ---------------------------------------------------------------------------
__global__ void k1_gray_yuv(const float* __restrict__ frame,
                            float* __restrict__ gray,
                            float* __restrict__ yuvsum,
                            __bf16* __restrict__ P) {
    const size_t i = (size_t)blockIdx.x * 256 + threadIdx.x;   // 0..B*HW-1
    const int b = (int)(i / HWPX);
    const int p = (int)(i % HWPX);
    const int y = p >> 9, x = p & 511;

    const float r  = frame[((size_t)b * 3 + 0) * HWPX + p];
    const float g  = frame[((size_t)b * 3 + 1) * HWPX + p];
    const float bl = frame[((size_t)b * 3 + 2) * HWPX + p];

    const float gy = 0.299f * r + 0.587f * g + 0.114f * bl;   // == Y channel
    gray[i] = gy;

    // padded bf16 conv input, channels 0..2 = RGB
    const size_t pb   = (size_t)b * 4 * WP2;
    const size_t pidx = (size_t)(y + 3) * WP + (x + 3);
    P[pb + 0 * WP2 + pidx] = (__bf16)r;
    P[pb + 1 * WP2 + pidx] = (__bf16)g;
    P[pb + 2 * WP2 + pidx] = (__bf16)bl;

    const float uu = -0.147f * r - 0.289f * g + 0.436f * bl;
    const float vv =  0.615f * r - 0.515f * g - 0.100f * bl;

    __shared__ float s0[256], s1[256], s2[256];
    const int t = threadIdx.x;
    s0[t] = gy; s1[t] = uu; s2[t] = vv;
    __syncthreads();
    for (int off = 128; off > 0; off >>= 1) {
        if (t < off) { s0[t] += s0[t + off]; s1[t] += s1[t + off]; s2[t] += s2[t + off]; }
        __syncthreads();
    }
    if (t == 0) {
        atomicAdd(&yuvsum[b * 3 + 0], s0[0]);
        atomicAdd(&yuvsum[b * 3 + 1], s1[0]);
        atomicAdd(&yuvsum[b * 3 + 2], s2[0]);
    }
}

// ---------------------------------------------------------------------------
// K2: uniform LBP code (unnormalized) into padded ch3 + per-image int max.
// Normalization by 1/max is folded into the conv filter (channel 3 taps).
// ---------------------------------------------------------------------------
__global__ void k2_lbp(const float* __restrict__ gray,
                       __bf16* __restrict__ P,
                       int* __restrict__ lbpmax) {
    const size_t i = (size_t)blockIdx.x * 256 + threadIdx.x;
    const int b = (int)(i / HWPX);
    const int p = (int)(i % HWPX);
    const int y = p >> 9, x = p & 511;
    const float* g = gray + (size_t)b * HWPX;
    const float c = g[p];

    const int dys[8] = {-1, -1, 0, 1, 1, 1, 0, -1};
    const int dxs[8] = { 0,  1, 1, 1, 0, -1, -1, -1};
    int bits[8], s = 0;
#pragma unroll
    for (int k = 0; k < 8; ++k) {
        const int yy = iclamp(y + dys[k], 0, HH - 1);
        const int xx = iclamp(x + dxs[k], 0, WW - 1);
        bits[k] = (g[yy * WW + xx] >= c) ? 1 : 0;
        s += bits[k];
    }
    int trans = 0;
#pragma unroll
    for (int k = 0; k < 8; ++k) {
        const int d = bits[k] - bits[(k + 7) & 7];
        trans += d < 0 ? -d : d;
    }
    const float code = (trans <= 2) ? (float)s : 9.0f;

    P[((size_t)b * 4 + 3) * WP2 + (size_t)(y + 3) * WP + (x + 3)] = (__bf16)code;

    __shared__ int sm[256];
    const int t = threadIdx.x;
    sm[t] = (int)code;
    __syncthreads();
    for (int off = 128; off > 0; off >>= 1) {
        if (t < off) sm[t] = sm[t] > sm[t + off] ? sm[t] : sm[t + off];
        __syncthreads();
    }
    if (t == 0) atomicMax(&lbpmax[b], sm[0]);
}

// ---------------------------------------------------------------------------
// K3: channel attention (3x3 MLP + sigmoid) and 1/max(lbp) per image.
// ---------------------------------------------------------------------------
__global__ void k3_finalize(const float* __restrict__ yuvsum,
                            const int* __restrict__ lbpmax,
                            const float* __restrict__ ca_w1,
                            const float* __restrict__ ca_w2,
                            float* __restrict__ scale,
                            float* __restrict__ invmx) {
    const int b = threadIdx.x;
    if (b >= BATCH) return;
    float pooled[3], h[3];
#pragma unroll
    for (int c = 0; c < 3; ++c) pooled[c] = yuvsum[b * 3 + c] * (1.0f / (float)HWPX);
#pragma unroll
    for (int j = 0; j < 3; ++j) {
        float a = 0.f;
#pragma unroll
        for (int k = 0; k < 3; ++k) a += pooled[k] * ca_w1[j * 3 + k];
        h[j] = a > 0.f ? a : 0.f;
    }
#pragma unroll
    for (int j = 0; j < 3; ++j) {
        float a = 0.f;
#pragma unroll
        for (int k = 0; k < 3; ++k) a += h[k] * ca_w2[j * 3 + k];
        scale[b * 3 + j] = sigmoidf(a);
    }
    const int mx = lbpmax[b] > 0 ? lbpmax[b] : 1;
    invmx[b] = 1.0f / (float)mx;
}

// ---------------------------------------------------------------------------
// K4: 7x7x4->1 'SAME' conv as implicit GEMM with v_wmma_f32_16x16x32_bf16.
//   Workgroup (256 thr = 8 waves) : 128 outputs in one row y of image b.
//   Each wave: 16 pixels (M), K = 196 padded to 224 -> 7 chained WMMAs.
//   K ordering: k = dy*28 + dx*4 + c.  Filter replicated over 16 N columns.
//   Patch (4ch x 7rows x 134cols bf16) staged into LDS by the TDM.
// ---------------------------------------------------------------------------
__global__ void k4_conv_wmma(const __bf16* __restrict__ P,
                             const float* __restrict__ sa_w,
                             const float* __restrict__ sa_b,
                             const float* __restrict__ invmx,
                             float* __restrict__ mask) {
    const int wg  = blockIdx.x;                 // 0 .. B*512*4-1
    const int b   = wg >> 11;                   // / (512*4)
    const int rem = wg & 2047;
    const int y   = rem >> 2;
    const int x0  = (rem & 3) * 128;

    __shared__ __bf16 patch[4 * 7 * 134];       // [c][dy][col], col in padded coords x0..x0+133
    __shared__ __bf16 filt[224];                // k = dy*28 + dx*4 + c (zero-padded)

    const int tid = threadIdx.x;
    const float inv3 = invmx[b];

    for (int k = tid; k < 224; k += 256) {
        float wv = 0.f;
        if (k < 196) {
            const int dy = k / 28, r = k % 28, dx = r >> 2, c = r & 3;
            wv = sa_w[c * 49 + dy * 7 + dx];
            if (c == 3) wv *= inv3;             // fold LBP normalization into filter
        }
        filt[k] = (__bf16)wv;
    }

#ifdef HAVE_TDM
    // --- Tensor Data Mover: one 3D tile (X=268B, Y=7 rows, Z=4 channels) ---
    if ((tid >> 5) == 0) {                      // wave 0 only (uniform branch)
        const unsigned long long gaddr =
            (unsigned long long)(const void*)(P + (((size_t)b * 4) * WP + y) * WP + x0);
        const unsigned int ldsoff =
            (unsigned int)(unsigned long long)(const void*)&patch[0];

        // D# group 0: count=1 | lds_addr | global_addr[56:0] | type=2
        u32x4 g0;
        g0[0] = 1u;
        g0[1] = ldsoff;
        g0[2] = (unsigned int)gaddr;
        g0[3] = ((unsigned int)((gaddr >> 32) & 0x01FFFFFFull)) | (2u << 30);

        // D# group 1: data_size=1B; tensor dims == tile dims (fully in-bounds);
        // tile = 268 x 7 x 4; row stride = WP*2 B; channel stride = WP2*2 B.
        const unsigned long long td0 = 268, td1 = 7;
        const unsigned long long tl0 = 268, tl1 = 7, tl2 = 4;
        const unsigned long long st0 = (unsigned long long)WP  * 2;   // 1036
        const unsigned long long st1 = (unsigned long long)WP2 * 2;   // 536648
        i32x8 g1;
        g1[0] = 0;                                                    // mask/flags/pad = 0
        g1[1] = (int)((td0 & 0xFFFFull) << 16);                       // [63:48] tensor_dim0 lo
        g1[2] = (int)(((td0 >> 16) & 0xFFFFull) | ((td1 & 0xFFFFull) << 16));
        g1[3] = (int)(((td1 >> 16) & 0xFFFFull) | ((tl0 & 0xFFFFull) << 16));
        g1[4] = (int)((tl1 & 0xFFFFull) | ((tl2 & 0xFFFFull) << 16));
        g1[5] = (int)(st0 & 0xFFFFFFFFull);
        g1[6] = (int)(((st0 >> 32) & 0xFFFFull) | ((st1 & 0xFFFFull) << 16));
        g1[7] = (int)((st1 >> 16) & 0xFFFFFFFFull);

        i32x4 g2;                                // tensor_dim2=4, tensor_dim3=1
        g2[0] = 4; g2[1] = 1; g2[2] = 0; g2[3] = 0;
        i32x4 g3;
        g3[0] = 0; g3[1] = 0; g3[2] = 0; g3[3] = 0;

#if __clang_major__ >= 23
        i32x8 gz;
        gz[0]=0; gz[1]=0; gz[2]=0; gz[3]=0; gz[4]=0; gz[5]=0; gz[6]=0; gz[7]=0;
        __builtin_amdgcn_tensor_load_to_lds(g0, g1, g2, g3, gz, 0);
#else
        __builtin_amdgcn_tensor_load_to_lds(g0, g1, g2, g3, 0);
#endif
        __builtin_amdgcn_s_wait_tensorcnt(0);   // transfer complete before barrier
    }
#else
    // --- fallback: cooperative staging ---
    for (int i = tid; i < 4 * 7 * 134; i += 256) {
        const int c   = i / (7 * 134);
        const int r2  = i % (7 * 134);
        const int dy  = r2 / 134;
        const int col = r2 % 134;
        patch[i] = P[(((size_t)b * 4 + c) * WP + (y + dy)) * WP + (x0 + col)];
    }
#endif
    __syncthreads();

    const int wave = tid >> 5;
    const int lane = tid & 31;
    const int m    = lane & 15;                 // A-matrix row held by this lane
    const int half = lane >> 4;
    const int xm   = wave * 16 + m;             // output x within the 128-tile

    v8f acc = {0.f, 0.f, 0.f, 0.f, 0.f, 0.f, 0.f, 0.f};
    for (int kc = 0; kc < 7; ++kc) {
        v16bf a, bm;
#pragma unroll
        for (int e = 0; e < 16; ++e) {
            // A fragment: 16-bit A 16x32 ISA layout -> K = (e<8?e:e+8) + half*8
            const int kA = kc * 32 + ((e < 8) ? e : e + 8) + half * 8;
            __bf16 av = (__bf16)0.0f;
            if (kA < 196) {
                const int dy = kA / 28, r = kA % 28, dx = r >> 2, c = r & 3;
                av = patch[c * (7 * 134) + dy * 134 + (xm + dx)];
            }
            a[e] = av;
            // B fragment: same K order as A -> K = (e<8?e:e+8) + half*8 (all N columns equal)
            bm[e] = filt[kc * 32 + ((e < 8) ? e : e + 8) + half * 8];
        }
        acc = __builtin_amdgcn_wmma_f32_16x16x32_bf16(
                  false, a, false, bm, (short)0, acc, false, false);
    }

    // Column N=0 of C lives in lanes 0 (M=0..7 across v0..v7) and 16 (M=8..15).
    const float bias = sa_b[0];
    if ((lane & 15) == 0) {
        const int mbase = half * 8;
#pragma unroll
        for (int r = 0; r < 8; ++r) {
            const float s = sigmoidf(acc[r] + bias);
            const int x = x0 + wave * 16 + mbase + r;
            mask[((size_t)b * HH + y) * WW + x] = s;
        }
    }
}

// ---------------------------------------------------------------------------
// K5: strength = 8x8 block mean of mask.
// ---------------------------------------------------------------------------
__global__ void k5_strength(const float* __restrict__ mask,
                            float* __restrict__ strength) {
    const int i = blockIdx.x * 256 + threadIdx.x;     // 0..B*64*64-1
    const int b  = i >> 12;
    const int rr = i & 4095;
    const int br = rr >> 6, bc = rr & 63;
    const float* mp = mask + (size_t)b * HWPX + (size_t)br * 8 * WW + bc * 8;
    float sum = 0.f;
#pragma unroll
    for (int iy = 0; iy < 8; ++iy)
#pragma unroll
        for (int ix = 0; ix < 8; ++ix) sum += mp[iy * WW + ix];
    strength[i] = sum * (1.0f / 64.0f);
}

// ---------------------------------------------------------------------------
// K6: fused channel-attention scale + rank-1 DCT watermark + YUV->RGB.
//   idct(dct(Y)+d*E43) == Y + d*D4[i]*D3[l]  (orthogonal DCT).
// ---------------------------------------------------------------------------
__global__ void k6_final(const float* __restrict__ frame,
                         const float* __restrict__ scale,
                         const float* __restrict__ strength,
                         const int* __restrict__ watermark,
                         float* __restrict__ out) {
    const size_t i = (size_t)blockIdx.x * 256 + threadIdx.x;
    const int b = (int)(i / HWPX);
    const int p = (int)(i % HWPX);
    const int y = p >> 9, x = p & 511;

    const float r  = frame[((size_t)b * 3 + 0) * HWPX + p];
    const float g  = frame[((size_t)b * 3 + 1) * HWPX + p];
    const float bl = frame[((size_t)b * 3 + 2) * HWPX + p];

    const float Y =  0.299f * r + 0.587f * g + 0.114f * bl;
    const float U = -0.147f * r - 0.289f * g + 0.436f * bl;
    const float V =  0.615f * r - 0.515f * g - 0.100f * bl;

    const float s0 = scale[b * 3 + 0];
    const float s1 = scale[b * 3 + 1];
    const float s2 = scale[b * 3 + 2];

    const int br = y >> 3, bc = x >> 3, ii = y & 7, ll = x & 7;
    const float st  = strength[(b << 12) + br * 64 + bc];
    const int bit   = watermark[bc & 63];             // idx[h,w] = (h*64+w)%64 = w
    const float dlt = ALPHA_C * st * (2.0f * (float)bit - 1.0f);

    const float Ym = Y * s0 + dlt * c_D4[ii] * c_D3[ll];
    const float Uu = U * s1;
    const float Vv = V * s2;

    out[((size_t)b * 3 + 0) * HWPX + p] = Ym + 1.140f * Vv;
    out[((size_t)b * 3 + 1) * HWPX + p] = Ym - 0.395f * Uu - 0.581f * Vv;
    out[((size_t)b * 3 + 2) * HWPX + p] = Ym + 2.032f * Uu;
}

// ---------------------------------------------------------------------------
// Launch
// ---------------------------------------------------------------------------
extern "C" void kernel_launch(void* const* d_in, const int* in_sizes, int n_in,
                              void* d_out, int out_size, void* d_ws, size_t ws_size,
                              hipStream_t stream) {
    const float* frame     = (const float*)d_in[0];
    const int*   watermark = (const int*)  d_in[1];
    const float* sa_w      = (const float*)d_in[2];
    const float* sa_b      = (const float*)d_in[3];
    const float* ca_w1     = (const float*)d_in[4];
    const float* ca_w2     = (const float*)d_in[5];

    char* ws = (char*)d_ws;
    // workspace layout (bytes)
    const size_t GRAY_OFF     = 0;                                   // B*HW*4
    const size_t YUVSUM_OFF   = GRAY_OFF + (size_t)BATCH * HWPX * 4; // 192
    const size_t LBPMAX_OFF   = YUVSUM_OFF + 256;                    // 64
    const size_t SCALE_OFF    = LBPMAX_OFF + 256;                    // 192
    const size_t INVMX_OFF    = SCALE_OFF + 256;                     // 64
    const size_t P_OFF        = ((INVMX_OFF + 256 + 255) / 256) * 256;
    const size_t P_BYTES      = (size_t)BATCH * 4 * WP2 * sizeof(__bf16);
    const size_t STRENGTH_OFF = P_OFF + ((P_BYTES + 255) / 256) * 256;

    float*  gray     = (float*)(ws + GRAY_OFF);
    float*  yuvsum   = (float*)(ws + YUVSUM_OFF);
    int*    lbpmax   = (int*)  (ws + LBPMAX_OFF);
    float*  scale    = (float*)(ws + SCALE_OFF);
    float*  invmx    = (float*)(ws + INVMX_OFF);
    __bf16* P        = (__bf16*)(ws + P_OFF);
    float*  strength = (float*)(ws + STRENGTH_OFF);

    float* out_rgb  = (float*)d_out;                               // B*3*H*W
    float* out_mask = (float*)d_out + (size_t)BATCH * 3 * HWPX;    // B*1*H*W

    // deterministic per-call init (graph-capture safe)
    hipMemsetAsync(yuvsum, 0, 192, stream);
    hipMemsetAsync(lbpmax, 0, 64, stream);
    hipMemsetAsync(P, 0, P_BYTES, stream);                          // zero pad border

    const int nPix = BATCH * HWPX;
    k1_gray_yuv  <<<nPix / 256, 256, 0, stream>>>(frame, gray, yuvsum, P);
    k2_lbp       <<<nPix / 256, 256, 0, stream>>>(gray, P, lbpmax);
    k3_finalize  <<<1, 32, 0, stream>>>(yuvsum, lbpmax, ca_w1, ca_w2, scale, invmx);
    k4_conv_wmma <<<BATCH * HH * (WW / 128), 256, 0, stream>>>(P, sa_w, sa_b, invmx, out_mask);
    k5_strength  <<<(BATCH * 64 * 64) / 256, 256, 0, stream>>>(out_mask, strength);
    k6_final     <<<nPix / 256, 256, 0, stream>>>(frame, scale, strength, watermark, out_rgb);

    (void)in_sizes; (void)n_in; (void)out_size; (void)ws_size;
}
